// SSMConnection_26560077758945
// MI455X (gfx1250) — matrix operand order repaired and checked by reading, not compiled, not run on hardware
//
#include <hip/hip_runtime.h>
#include <hip/hip_bf16.h>

// ---------------- types ----------------
typedef __attribute__((ext_vector_type(16))) __bf16 v16bf;
typedef __attribute__((ext_vector_type(8)))  __bf16 v8bf;
typedef __attribute__((ext_vector_type(8)))  float  v8f;

#define HID    2048
#define SELN   512
#define RESN   1536
#define YSZ    2048
#define RANK   64
#define CATN   4096
#define ROWS   32          // tokens per workgroup
#define CAT_STRIDE 4104    // padded ushort stride (bank-conflict-free A reads)
#define LOW_STRIDE 72

__device__ __forceinline__ unsigned short f2bf(float f) {
    unsigned int u = __float_as_uint(f);
    u += 0x7fffu + ((u >> 16) & 1u);   // round-to-nearest-even
    return (unsigned short)(u >> 16);
}

__device__ __forceinline__ uint2 pack4bf(float a, float b, float c, float d) {
    uint2 r;
    r.x = (unsigned)f2bf(a) | ((unsigned)f2bf(b) << 16);
    r.y = (unsigned)f2bf(c) | ((unsigned)f2bf(d) << 16);
    return r;
}

__device__ __forceinline__ float softplusf(float x) {
    return (x > 20.f) ? x : log1pf(expf(x));
}

// A-matrix tile (16x32 bf16) from LDS. Lane = M row; elems 0..7 = K(kb..kb+7),
// elems 8..15 = K(kb+16..kb+23), kb = (lane<16)?0:8  (per ISA A 16-bit layout).
__device__ __forceinline__ v16bf loadA_lds(const unsigned short* base, int m0,
                                           int k0, int lane, int stride) {
    int m  = m0 + (lane & 15);
    int kb = (lane < 16) ? 0 : 8;
    const __bf16* p = (const __bf16*)(base + (size_t)m * stride + k0 + kb);
    v8bf lo = *(const v8bf*)(p);
    v8bf hi = *(const v8bf*)(p + 16);
    return __builtin_shufflevector(lo, hi, 0,1,2,3,4,5,6,7,8,9,10,11,12,13,14,15);
}

// B-matrix tile (32x16 bf16) from global, where B[k][n] = M[n][k] (row-major M,
// row n, stride `stride`). Lane = N col; elems 0..7 = K(kb..kb+7),
// elems 8..15 = K(kb+16..kb+23), kb = (lane<16)?0:8 (matches A layout).
__device__ __forceinline__ v16bf loadB_g(const unsigned short* base, int n0,
                                         int k0, int lane, int stride) {
    int n    = n0 + (lane & 15);
    int koff = (lane < 16) ? 0 : 8;
    const __bf16* p = (const __bf16*)(base + (size_t)n * stride + k0 + koff);
    v8bf lo = *(const v8bf*)(p);
    v8bf hi = *(const v8bf*)(p + 16);
    return __builtin_shufflevector(lo, hi, 0,1,2,3,4,5,6,7,8,9,10,11,12,13,14,15);
}

__device__ __forceinline__ v8f bfwmma(v16bf a, v16bf b, v8f c) {
    return __builtin_amdgcn_wmma_f32_16x16x32_bf16(
        /*neg_a=*/false, a, /*neg_b=*/false, b,
        /*c_mod=*/(short)0, c, /*reuse_a=*/false, /*reuse_b=*/false);
}

// ---------------- weight f32 -> bf16 conversion ----------------
__global__ void cvt_bf16_kernel(const float* __restrict__ src,
                                unsigned short* __restrict__ dst, int n) {
    int i = blockIdx.x * blockDim.x + threadIdx.x;
    if (i < n) dst[i] = f2bf(src[i]);
}

// ---------------- fused main kernel ----------------
extern __shared__ char smem_raw[];

__global__ __launch_bounds__(256, 1)
void ssm_fused_kernel(const float* __restrict__ hid,
                      const float* __restrict__ y,
                      const float* __restrict__ Avec,
                      const float* __restrict__ du_b,
                      const float* __restrict__ sel_gamma,
                      const float* __restrict__ sel_beta,
                      const float* __restrict__ res_gamma,
                      const float* __restrict__ res_beta,
                      const unsigned short* __restrict__ Wbf,   // [2048][2048]
                      const unsigned short* __restrict__ ddbf,  // [64][4096]
                      const unsigned short* __restrict__ dubf,  // [512][64]
                      float* __restrict__ out) {
    unsigned short* cat  = (unsigned short*)smem_raw;              // [ROWS][CAT_STRIDE]
    unsigned short* lowb = cat + (size_t)ROWS * CAT_STRIDE;        // [ROWS][LOW_STRIDE]

    const int wave = threadIdx.x >> 5;
    const int lane = threadIdx.x & 31;
    const size_t row0 = (size_t)blockIdx.x * ROWS;

    const v8f vzero = {0.f,0.f,0.f,0.f,0.f,0.f,0.f,0.f};

    // ---- Phase 1: LayerNorm + pack cat = [normed(2048) | y(2048)] as bf16 ----
    // Each lane owns contiguous column chunks -> float4 loads, b64 LDS stores.
    for (int rr = 0; rr < ROWS / 8; ++rr) {
        const int r = wave * (ROWS / 8) + rr;
        const float4* h4 = (const float4*)(hid + (row0 + r) * HID);
        const float4* y4 = (const float4*)(y   + (row0 + r) * YSZ);
        unsigned short* crow = cat + (size_t)r * CAT_STRIDE;

        // --- sel: lane owns columns [lane*16, lane*16+16) ---
        float4 xs[4];
        float s = 0.f, s2 = 0.f;
        #pragma unroll
        for (int i = 0; i < 4; ++i) {
            float4 v = h4[lane * 4 + i];
            xs[i] = v;
            s  += v.x + v.y + v.z + v.w;
            s2 += v.x*v.x + v.y*v.y + v.z*v.z + v.w*v.w;
        }
        #pragma unroll
        for (int off = 16; off; off >>= 1) {
            s  += __shfl_xor(s,  off);
            s2 += __shfl_xor(s2, off);
        }
        float mu   = s * (1.f / 512.f);
        float var  = s2 * (1.f / 512.f) - mu * mu;
        float rstd = rsqrtf(var + 1e-5f);
        #pragma unroll
        for (int i = 0; i < 4; ++i) {
            int c = lane * 16 + i * 4;
            float4 g  = *(const float4*)(sel_gamma + c);
            float4 be = *(const float4*)(sel_beta  + c);
            *(uint2*)(crow + c) = pack4bf(
                (xs[i].x - mu) * rstd * g.x + be.x,
                (xs[i].y - mu) * rstd * g.y + be.y,
                (xs[i].z - mu) * rstd * g.z + be.z,
                (xs[i].w - mu) * rstd * g.w + be.w);
        }

        // --- res: lane owns columns [lane*48, lane*48+48) of the RES block ---
        float4 xr[12];
        s = 0.f; s2 = 0.f;
        #pragma unroll
        for (int i = 0; i < 12; ++i) {
            float4 v = h4[(SELN / 4) + lane * 12 + i];
            xr[i] = v;
            s  += v.x + v.y + v.z + v.w;
            s2 += v.x*v.x + v.y*v.y + v.z*v.z + v.w*v.w;
        }
        #pragma unroll
        for (int off = 16; off; off >>= 1) {
            s  += __shfl_xor(s,  off);
            s2 += __shfl_xor(s2, off);
        }
        mu   = s * (1.f / 1536.f);
        var  = s2 * (1.f / 1536.f) - mu * mu;
        rstd = rsqrtf(var + 1e-5f);
        #pragma unroll
        for (int i = 0; i < 12; ++i) {
            int c = lane * 48 + i * 4;               // column within RES block
            float4 g  = *(const float4*)(res_gamma + c);
            float4 be = *(const float4*)(res_beta  + c);
            *(uint2*)(crow + SELN + c) = pack4bf(
                (xr[i].x - mu) * rstd * g.x + be.x,
                (xr[i].y - mu) * rstd * g.y + be.y,
                (xr[i].z - mu) * rstd * g.z + be.z,
                (xr[i].w - mu) * rstd * g.w + be.w);
        }

        // --- y: lane owns columns [lane*64, lane*64+64) ---
        #pragma unroll
        for (int i = 0; i < 16; ++i) {
            float4 v = y4[lane * 16 + i];
            *(uint2*)(crow + HID + lane * 64 + i * 4) =
                pack4bf(v.x, v.y, v.z, v.w);
        }
    }
    __syncthreads();

    // ---- Phase 2: low[32x64] = cat[32x4096] x dd^T ; one 16x16 tile / wave ----
    {
        const int mt = wave >> 2, nt = wave & 3;
        v8f acc = vzero;
        for (int k0 = 0; k0 < CATN; k0 += 32) {
            v16bf a = loadA_lds(cat, mt * 16, k0, lane, CAT_STRIDE);
            v16bf b = loadB_g(ddbf, nt * 16, k0, lane, CATN);
            acc = bfwmma(a, b, acc);
        }
        const int n  = nt * 16 + (lane & 15);
        const int mb = mt * 16 + ((lane < 16) ? 0 : 8);
        #pragma unroll
        for (int v = 0; v < 8; ++v)
            lowb[(size_t)(mb + v) * LOW_STRIDE + n] = f2bf(acc[v]);
    }
    __syncthreads();

    // ---- Phase 3: delta -> A_bar for this wave's SEL strip [wave*64, wave*64+64) ----
    float abar[8][8];   // [mt*4+nt][v]
    float aneg[4];      // per n-tile, value for this lane's column
    {
        #pragma unroll
        for (int nt = 0; nt < 4; ++nt) {
            int n = wave * 64 + nt * 16 + (lane & 15);
            aneg[nt] = -softplusf(Avec[n]);
        }
        #pragma unroll
        for (int mt = 0; mt < 2; ++mt) {
            #pragma unroll
            for (int nt = 0; nt < 4; ++nt) {
                v8f acc = vzero;
                #pragma unroll
                for (int k0 = 0; k0 < RANK; k0 += 32) {
                    v16bf a = loadA_lds(lowb, mt * 16, k0, lane, LOW_STRIDE);
                    v16bf b = loadB_g(dubf, wave * 64 + nt * 16, k0, lane, RANK);
                    acc = bfwmma(a, b, acc);
                }
                int n = wave * 64 + nt * 16 + (lane & 15);
                float bias = du_b[n];
                #pragma unroll
                for (int v = 0; v < 8; ++v) {
                    float d = softplusf(acc[v] + bias);
                    abar[mt * 4 + nt][v] = expf(d * aneg[nt]);
                }
            }
        }
    }

    // ---- Phase 4: wy over SEL strip + fused combine -> out[:, 0:512] ----
    {
        v8f acc[8];
        #pragma unroll
        for (int t = 0; t < 8; ++t) acc[t] = vzero;
        for (int k0 = 0; k0 < YSZ; k0 += 32) {
            v16bf a0 = loadA_lds(cat, 0,  HID + k0, lane, CAT_STRIDE);
            v16bf a1 = loadA_lds(cat, 16, HID + k0, lane, CAT_STRIDE);
            #pragma unroll
            for (int nt = 0; nt < 4; ++nt) {
                v16bf b = loadB_g(Wbf, wave * 64 + nt * 16, k0, lane, YSZ);
                acc[0 * 4 + nt] = bfwmma(a0, b, acc[0 * 4 + nt]);
                acc[1 * 4 + nt] = bfwmma(a1, b, acc[1 * 4 + nt]);
            }
        }
        #pragma unroll
        for (int mt = 0; mt < 2; ++mt) {
            #pragma unroll
            for (int nt = 0; nt < 4; ++nt) {
                const int t  = mt * 4 + nt;
                const int n  = wave * 64 + nt * 16 + (lane & 15);
                const int mb = mt * 16 + ((lane < 16) ? 0 : 8);
                const float an = aneg[nt];
                #pragma unroll
                for (int v = 0; v < 8; ++v) {
                    size_t row = row0 + mb + v;
                    float selh = hid[row * HID + n];
                    float ab   = abar[t][v];
                    float bb   = (ab - 1.0f) / an;
                    out[row * HID + n] = ab * selh + bb * acc[t][v];
                }
            }
        }
    }

    // ---- Phase 5: wy over RES strips + fused residual add -> out[:, 512:2048] ----
    for (int c = 0; c < 3; ++c) {
        const int nbase = SELN + wave * 192 + c * 64;
        v8f acc[8];
        #pragma unroll
        for (int t = 0; t < 8; ++t) acc[t] = vzero;
        for (int k0 = 0; k0 < YSZ; k0 += 32) {
            v16bf a0 = loadA_lds(cat, 0,  HID + k0, lane, CAT_STRIDE);
            v16bf a1 = loadA_lds(cat, 16, HID + k0, lane, CAT_STRIDE);
            #pragma unroll
            for (int nt = 0; nt < 4; ++nt) {
                v16bf b = loadB_g(Wbf, nbase + nt * 16, k0, lane, YSZ);
                acc[0 * 4 + nt] = bfwmma(a0, b, acc[0 * 4 + nt]);
                acc[1 * 4 + nt] = bfwmma(a1, b, acc[1 * 4 + nt]);
            }
        }
        #pragma unroll
        for (int mt = 0; mt < 2; ++mt) {
            #pragma unroll
            for (int nt = 0; nt < 4; ++nt) {
                const int t  = mt * 4 + nt;
                const int n  = nbase + nt * 16 + (lane & 15);
                const int mb = mt * 16 + ((lane < 16) ? 0 : 8);
                #pragma unroll
                for (int v = 0; v < 8; ++v) {
                    size_t row = row0 + mb + v;
                    out[row * HID + n] = hid[row * HID + n] + acc[t][v];
                }
            }
        }
    }
}

// ---------------- launcher ----------------
extern "C" void kernel_launch(void* const* d_in, const int* in_sizes, int n_in,
                              void* d_out, int out_size, void* d_ws, size_t ws_size,
                              hipStream_t stream) {
    const float* hid = (const float*)d_in[0];
    const float* y   = (const float*)d_in[1];
    const float* A   = (const float*)d_in[2];
    const float* W   = (const float*)d_in[3];   // [2048][2048] (h, y)
    const float* dd  = (const float*)d_in[4];   // [64][4096]
    const float* du  = (const float*)d_in[5];   // [512][64]
    const float* dub = (const float*)d_in[6];
    const float* sg  = (const float*)d_in[7];
    const float* sb  = (const float*)d_in[8];
    const float* rg  = (const float*)d_in[9];
    const float* rb  = (const float*)d_in[10];
    float* out = (float*)d_out;

    unsigned short* Wbf  = (unsigned short*)d_ws;            // 2048*2048
    unsigned short* ddbf = Wbf  + (size_t)HID * YSZ;         // 64*4096
    unsigned short* dubf = ddbf + (size_t)RANK * CATN;       // 512*64

    const int nW  = HID * YSZ;
    const int nDD = RANK * CATN;
    const int nDU = SELN * RANK;
    cvt_bf16_kernel<<<(nW  + 255) / 256, 256, 0, stream>>>(W,  Wbf,  nW);
    cvt_bf16_kernel<<<(nDD + 255) / 256, 256, 0, stream>>>(dd, ddbf, nDD);
    cvt_bf16_kernel<<<(nDU + 255) / 256, 256, 0, stream>>>(du, dubf, nDU);

    const int tokens = in_sizes[0] / HID;   // 16384
    const size_t smem =
        ((size_t)ROWS * CAT_STRIDE + (size_t)ROWS * LOW_STRIDE) * sizeof(unsigned short);
    ssm_fused_kernel<<<tokens / ROWS, 256, smem, stream>>>(
        hid, y, A, dub, sg, sb, rg, rb, Wbf, ddbf, dubf, out);
}
